// PDFConv_73632919322719
// MI455X (gfx1250) — matrix operand order repaired and checked by reading, not compiled, not run on hardware
//
#include <hip/hip_runtime.h>
#include <hip/hip_bf16.h>

#define HDIM 128
#define BN_EPS 1e-5f

typedef __attribute__((ext_vector_type(16))) __bf16 v16bf;
typedef __attribute__((ext_vector_type(8)))  float  v8f;

union BF16x16 { v16bf v; uint4 q[2]; };

// Branchless exact-GELU: erf via Abramowitz–Stegun 7.1.26 (|err|<1.5e-7),
// using hardware v_rcp_f32 / v_exp_f32 — no divergent range splits.
__device__ __forceinline__ float gelu_exact(float x) {
    const float s = x * 0.70710678118654752f;      // x / sqrt(2)
    const float a = __builtin_fabsf(s);
    const float t = __builtin_amdgcn_rcpf(__builtin_fmaf(0.3275911f, a, 1.0f));
    float p = __builtin_fmaf(t, 1.061405429f, -1.453152027f);
    p = __builtin_fmaf(t, p, 1.421413741f);
    p = __builtin_fmaf(t, p, -0.284496736f);
    p = __builtin_fmaf(t, p, 0.254829592f);
    p *= t;
    const float e = __builtin_amdgcn_exp2f(-a * a * 1.4426950408889634f);
    const float erf_a = __builtin_fmaf(-p, e, 1.0f);
    const float erf_s = __builtin_copysignf(erf_a, s);
    return 0.5f * x * (1.0f + erf_s);
}

// ---------------------------------------------------------------------------
// Kernel 0: per-call init — zero aggr + stats, convert weights f32 -> bf16
// ---------------------------------------------------------------------------
__global__ void prep_kernel(const float* __restrict__ W1,
                            const float* __restrict__ W2,
                            const float* __restrict__ W3,
                            __bf16* __restrict__ W1b,
                            __bf16* __restrict__ W2b,
                            __bf16* __restrict__ W3b,
                            float* __restrict__ aggr,
                            float* __restrict__ stats,
                            int aggr_elems) {
    const int tid    = blockIdx.x * blockDim.x + threadIdx.x;
    const int stride = gridDim.x * blockDim.x;
    for (int i = tid; i < aggr_elems; i += stride) aggr[i] = 0.0f;
    for (int i = tid; i < 8 * HDIM;   i += stride) stats[i] = 0.0f;
    for (int i = tid; i < HDIM * HDIM; i += stride) {
        W1b[i] = (__bf16)W1[i];
        W2b[i] = (__bf16)W2[i];
        W3b[i] = (__bf16)W3[i];
    }
}

// ---------------------------------------------------------------------------
// Kernel 1: edge phase.
// Tile of 128 edges: A = bf16(x_feat[src] + edge_attr)  [128 x 128]
// v = gelu(A @ W1^T + b1) * bases ; atomic scatter-add into aggr[dst].
// ---------------------------------------------------------------------------
__launch_bounds__(256, 2)
__global__ void edge_kernel(const float* __restrict__ x_feat,
                            const float* __restrict__ edge_attr,
                            const float* __restrict__ bases,
                            const int*   __restrict__ src,
                            const int*   __restrict__ dst,
                            const __bf16* __restrict__ W1b,
                            const float* __restrict__ b1,
                            float* __restrict__ aggr,
                            int E) {
    __shared__ __align__(16) __bf16 sA[128][136];   // 34.8 KB, padded stride

    const int tid   = threadIdx.x;
    const int tileE = blockIdx.x * 128;
    const bool full = (tileE + 128 <= E);           // uniform per block

    // Prefetch next tile of the streaming operands into L2 (global_prefetch_b8)
    if (tileE + 128 < E) {
        const size_t nb = (size_t)(tileE + 128) * HDIM + (size_t)tid * 64;
        __builtin_prefetch(&edge_attr[nb], 0, 0);
        __builtin_prefetch(&bases[nb], 0, 0);
    }

    // Build A tile: 4096 float4 chunks / 256 threads = 16 each (coalesced).
    if (full) {
        for (int c = tid; c < 4096; c += 256) {
            const int e  = c >> 5;
            const int k4 = (c & 31) << 2;
            const int eg = tileE + e;
            const int sr = src[eg];
            const float4 ea = *(const float4*)&edge_attr[(size_t)eg * HDIM + k4];
            const float4 xf = *(const float4*)&x_feat[(size_t)sr * HDIM + k4];
            union { __bf16 h[4]; uint2 u; } p;
            p.h[0] = (__bf16)(ea.x + xf.x);
            p.h[1] = (__bf16)(ea.y + xf.y);
            p.h[2] = (__bf16)(ea.z + xf.z);
            p.h[3] = (__bf16)(ea.w + xf.w);
            *(uint2*)&sA[e][k4] = p.u;
        }
    } else {
        for (int c = tid; c < 4096; c += 256) {
            const int e  = c >> 5;
            const int k4 = (c & 31) << 2;
            const int eg = tileE + e;
            union { __bf16 h[4]; uint2 u; } p;
            p.u = make_uint2(0u, 0u);
            if (eg < E) {
                const int sr = src[eg];
                const float4 ea = *(const float4*)&edge_attr[(size_t)eg * HDIM + k4];
                const float4 xf = *(const float4*)&x_feat[(size_t)sr * HDIM + k4];
                p.h[0] = (__bf16)(ea.x + xf.x);
                p.h[1] = (__bf16)(ea.y + xf.y);
                p.h[2] = (__bf16)(ea.z + xf.z);
                p.h[3] = (__bf16)(ea.w + xf.w);
            }
            *(uint2*)&sA[e][k4] = p.u;
        }
    }
    __syncthreads();

    const int wave = tid >> 5;
    const int lane = tid & 31;
    const int lhi  = lane >> 4;      // half-wave select
    const int llo  = lane & 15;
    const int mrow = wave * 16;

    // A fragments (16-bit 16x32 layout): runs [kb,kb+8) and [kb+16,kb+24)
    BF16x16 afrag[4];
#pragma unroll
    for (int ks = 0; ks < 4; ++ks) {
        const __bf16* rp = &sA[mrow + llo][ks * 32 + lhi * 8];
        afrag[ks].q[0] = *(const uint4*)rp;
        afrag[ks].q[1] = *(const uint4*)(rp + 16);
    }

    // Edge ids / dst rows for this wave's strip (C layout: m = v + 8*lhi)
    int eg[8], dr[8];
#pragma unroll
    for (int v = 0; v < 8; ++v) {
        eg[v] = tileE + mrow + v + lhi * 8;
        dr[v] = (full || eg[v] < E) ? dst[eg[v]] : 0;
    }

#pragma unroll
    for (int nt = 0; nt < 8; ++nt) {
        const int nn = nt * 16 + llo;
        v8f c = {};
#pragma unroll
        for (int ks = 0; ks < 4; ++ks) {
            // B = W1^T : B[k][n] = W1[n][k]; K runs [kb,kb+8) and [kb+16,kb+24) per lane from
            // global bf16 weights (32 KB, WGP$/L2 resident)
            BF16x16 bfrag;
            const __bf16* wp = W1b + (size_t)nn * HDIM + ks * 32 + lhi * 8;
            bfrag.q[0] = *(const uint4*)wp;
            bfrag.q[1] = *(const uint4*)(wp + 16);
            c = __builtin_amdgcn_wmma_f32_16x16x32_bf16(
                    false, afrag[ks].v, false, bfrag.v, (short)0, c, false, false);
        }
        const float bias = b1[nn];
        if (full) {
#pragma unroll
            for (int v = 0; v < 8; ++v) {
                const float g   = gelu_exact(c[v] + bias);
                const float val = g * bases[(size_t)eg[v] * HDIM + nn];
                atomicAdd(&aggr[(size_t)dr[v] * HDIM + nn], val);
            }
        } else {
#pragma unroll
            for (int v = 0; v < 8; ++v) {
                if (eg[v] < E) {
                    const float g   = gelu_exact(c[v] + bias);
                    const float val = g * bases[(size_t)eg[v] * HDIM + nn];
                    atomicAdd(&aggr[(size_t)dr[v] * HDIM + nn], val);
                }
            }
        }
    }
}

// ---------------------------------------------------------------------------
// Kernel 2/3: node FFN GEMMs.
// TRANSFORM=false: Y = Ain @ W^T + bias                    (Ain = aggr)
// TRANSFORM=true : Y = gelu(scale*Ain + shift) @ W^T + bias (BN+GELU fused
//                  into the A-load path)
// Per-column sum / sum-of-squares of Y reduced through LDS, then atomics.
// ---------------------------------------------------------------------------
template <bool TRANSFORM>
__launch_bounds__(256, 2)
__global__ void node_gemm_kernel(const float* __restrict__ Ain,
                                 const __bf16* __restrict__ Wb,
                                 const float* __restrict__ bias,
                                 const float* __restrict__ scale,
                                 const float* __restrict__ shift,
                                 float* __restrict__ Yout,
                                 float* __restrict__ ssum,
                                 float* __restrict__ ssq,
                                 int nrows) {
    __shared__ __align__(16) __bf16 sA[128][136];
    __shared__ float redS[HDIM];
    __shared__ float redQ[HDIM];

    const int tid   = threadIdx.x;
    const int tileR = blockIdx.x * 128;
    const bool full = (tileR + 128 <= nrows);       // uniform per block

    if (tid < HDIM) { redS[tid] = 0.0f; redQ[tid] = 0.0f; }

    for (int c = tid; c < 4096; c += 256) {
        const int r  = c >> 5;
        const int k4 = (c & 31) << 2;
        const int rg = tileR + r;
        union { __bf16 h[4]; uint2 u; } p;
        p.u = make_uint2(0u, 0u);
        if (full || rg < nrows) {
            float4 a = *(const float4*)&Ain[(size_t)rg * HDIM + k4];
            if constexpr (TRANSFORM) {
                const float4 sc = *(const float4*)&scale[k4];
                const float4 sh = *(const float4*)&shift[k4];
                a.x = gelu_exact(__builtin_fmaf(sc.x, a.x, sh.x));
                a.y = gelu_exact(__builtin_fmaf(sc.y, a.y, sh.y));
                a.z = gelu_exact(__builtin_fmaf(sc.z, a.z, sh.z));
                a.w = gelu_exact(__builtin_fmaf(sc.w, a.w, sh.w));
            }
            p.h[0] = (__bf16)a.x; p.h[1] = (__bf16)a.y;
            p.h[2] = (__bf16)a.z; p.h[3] = (__bf16)a.w;
        }
        *(uint2*)&sA[r][k4] = p.u;
    }
    __syncthreads();

    const int wave = tid >> 5;
    const int lane = tid & 31;
    const int lhi  = lane >> 4;
    const int llo  = lane & 15;
    const int mrow = wave * 16;

    BF16x16 afrag[4];
#pragma unroll
    for (int ks = 0; ks < 4; ++ks) {
        const __bf16* rp = &sA[mrow + llo][ks * 32 + lhi * 8];
        afrag[ks].q[0] = *(const uint4*)rp;
        afrag[ks].q[1] = *(const uint4*)(rp + 16);
    }

#pragma unroll
    for (int nt = 0; nt < 8; ++nt) {
        const int nn = nt * 16 + llo;
        v8f c = {};
#pragma unroll
        for (int ks = 0; ks < 4; ++ks) {
            BF16x16 bfrag;
            const __bf16* wp = Wb + (size_t)nn * HDIM + ks * 32 + lhi * 8;
            bfrag.q[0] = *(const uint4*)wp;
            bfrag.q[1] = *(const uint4*)(wp + 16);
            c = __builtin_amdgcn_wmma_f32_16x16x32_bf16(
                    false, afrag[ks].v, false, bfrag.v, (short)0, c, false, false);
        }
        const float b = bias[nn];
        float s = 0.0f, q = 0.0f;
        if (full) {
#pragma unroll
            for (int v = 0; v < 8; ++v) {
                const int row = tileR + mrow + v + lhi * 8;
                const float y = c[v] + b;
                Yout[(size_t)row * HDIM + nn] = y;
                s += y;
                q += y * y;
            }
        } else {
#pragma unroll
            for (int v = 0; v < 8; ++v) {
                const int row = tileR + mrow + v + lhi * 8;
                if (row < nrows) {
                    const float y = c[v] + b;
                    Yout[(size_t)row * HDIM + nn] = y;
                    s += y;
                    q += y * y;
                }
            }
        }
        // lanes l and l^16 hold the same column nn with disjoint rows
        s += __shfl_xor(s, 16, 32);
        q += __shfl_xor(q, 16, 32);
        if (lhi == 0) {
            atomicAdd(&redS[nn], s);      // ds_add_f32
            atomicAdd(&redQ[nn], q);
        }
    }
    __syncthreads();
    if (tid < HDIM) {
        atomicAdd(&ssum[tid], redS[tid]);
        atomicAdd(&ssq[tid],  redQ[tid]);
    }
}

// ---------------------------------------------------------------------------
// Fold BN stats into per-column affine scale/shift
// ---------------------------------------------------------------------------
__global__ void bn_finalize_kernel(const float* __restrict__ ssum,
                                   const float* __restrict__ ssq,
                                   const float* __restrict__ gamma,
                                   const float* __restrict__ beta,
                                   float* __restrict__ scale,
                                   float* __restrict__ shift,
                                   float inv_n) {
    const int t = threadIdx.x;
    if (t < HDIM) {
        const float mu  = ssum[t] * inv_n;
        const float var = ssq[t] * inv_n - mu * mu;
        const float a   = gamma[t] * rsqrtf(var + BN_EPS);
        scale[t] = a;
        shift[t] = beta[t] - mu * a;
    }
}

// ---------------------------------------------------------------------------
// Final BN2 + GELU, in place on d_out (vectorized float4)
// ---------------------------------------------------------------------------
__global__ void final_gelu_kernel(float* __restrict__ out,
                                  const float* __restrict__ scale,
                                  const float* __restrict__ shift,
                                  int total4) {
    const int i = blockIdx.x * blockDim.x + threadIdx.x;
    if (i < total4) {
        const int k4 = (i & 31) << 2;                 // column group
        float4 y = *(float4*)&out[(size_t)i * 4];
        const float4 sc = *(const float4*)&scale[k4];
        const float4 sh = *(const float4*)&shift[k4];
        y.x = gelu_exact(__builtin_fmaf(sc.x, y.x, sh.x));
        y.y = gelu_exact(__builtin_fmaf(sc.y, y.y, sh.y));
        y.z = gelu_exact(__builtin_fmaf(sc.z, y.z, sh.z));
        y.w = gelu_exact(__builtin_fmaf(sc.w, y.w, sh.w));
        *(float4*)&out[(size_t)i * 4] = y;
    }
}

// ---------------------------------------------------------------------------
extern "C" void kernel_launch(void* const* d_in, const int* in_sizes, int n_in,
                              void* d_out, int out_size, void* d_ws, size_t ws_size,
                              hipStream_t stream) {
    const float* x_feat    = (const float*)d_in[0];
    const float* edge_attr = (const float*)d_in[1];
    const float* bases     = (const float*)d_in[2];
    const int*   src       = (const int*)d_in[3];
    const int*   dst       = (const int*)d_in[4];
    const float* W1  = (const float*)d_in[5];
    const float* b1  = (const float*)d_in[6];
    const float* W2  = (const float*)d_in[7];
    const float* b2  = (const float*)d_in[8];
    const float* g1  = (const float*)d_in[9];
    const float* be1 = (const float*)d_in[10];
    const float* W3  = (const float*)d_in[11];
    const float* b3  = (const float*)d_in[12];
    const float* g2  = (const float*)d_in[13];
    const float* be2 = (const float*)d_in[14];

    const int N = in_sizes[0] / HDIM;
    const int E = in_sizes[3];

    // workspace layout
    char* ws = (char*)d_ws;
    __bf16* W1b  = (__bf16*)(ws);
    __bf16* W2b  = (__bf16*)(ws + 32768);
    __bf16* W3b  = (__bf16*)(ws + 65536);
    float*  stats = (float*)(ws + 98304);            // 8 * 128 floats
    float*  sum1 = stats,        *ssq1 = stats + 128;
    float*  sum2 = stats + 256,  *ssq2 = stats + 384;
    float*  scale1 = stats + 512, *shift1 = stats + 640;
    float*  scale2 = stats + 768, *shift2 = stats + 896;
    float*  aggr = (float*)(ws + 102400);
    float*  y1   = (float*)(ws + 102400 + (size_t)N * HDIM * sizeof(float));

    float* out = (float*)d_out;

    // 0) init (zeroes everything this launch depends on -> graph-replay safe)
    prep_kernel<<<2048, 256, 0, stream>>>(W1, W2, W3, W1b, W2b, W3b,
                                          aggr, stats, N * HDIM);

    // 1) edge GEMM + gelu*bases + scatter-add
    edge_kernel<<<(E + 127) / 128, 256, 0, stream>>>(
        x_feat, edge_attr, bases, src, dst, W1b, b1, aggr, E);

    const int nodeBlocks = (N + 127) / 128;

    // 2) y1 = aggr @ W2^T + b2 ; accumulate BN1 stats
    node_gemm_kernel<false><<<nodeBlocks, 256, 0, stream>>>(
        aggr, W2b, b2, nullptr, nullptr, y1, sum1, ssq1, N);

    // 3) BN1 -> affine
    bn_finalize_kernel<<<1, 128, 0, stream>>>(sum1, ssq1, g1, be1,
                                              scale1, shift1, 1.0f / (float)N);

    // 4) y2 = gelu(bn1(y1)) @ W3^T + b3 ; accumulate BN2 stats ; y2 -> d_out
    node_gemm_kernel<true><<<nodeBlocks, 256, 0, stream>>>(
        y1, W3b, b3, scale1, shift1, out, sum2, ssq2, N);

    // 5) BN2 -> affine
    bn_finalize_kernel<<<1, 128, 0, stream>>>(sum2, ssq2, g2, be2,
                                              scale2, shift2, 1.0f / (float)N);

    // 6) out = gelu(bn2(y2)) in place (N*HDIM is a multiple of 4)
    const int total4 = (N * HDIM) / 4;
    final_gelu_kernel<<<(total4 + 255) / 256, 256, 0, stream>>>(
        out, scale2, shift2, total4);
}
